// MultiHeadSelfAttention_49417893708011
// MI455X (gfx1250) — matrix operand order repaired and checked by reading, not compiled, not run on hardware
//
#include <hip/hip_runtime.h>
#include <hip/hip_bf16.h>

#define D_MODEL 1024
#define NUM_HEADS 16
#define DEPTH 64
#define BATCH 2
#define SEQ 2048
#define NEG_INF -1e9f

typedef __attribute__((ext_vector_type(16))) _Float16 v16h;
typedef __attribute__((ext_vector_type(8)))  float    v8f;

// ---------------------------------------------------------------------------
// WMMA fragment helpers (CDNA5 wave32 layouts, CDNA5 ISA §7.12.2)
// ---------------------------------------------------------------------------

// A fragment: 16x32 f16 tile, lane holds row M = rowBase + (lane&15),
// K striping: VGPR0..3 -> K = 8g+2j+h, VGPR4..7 -> K = 16+8g+2(j-4)+h.
// Two contiguous 8-half runs per lane -> compiler emits 2x ds_load_b128.
__device__ __forceinline__ v16h frag_a(const _Float16* sh, int ldStride,
                                       int rowBase, int kBase, int lane) {
  const int g = lane >> 4;
  const _Float16* p = sh + (size_t)(rowBase + (lane & 15)) * ldStride + kBase;
  v16h a;
#pragma unroll
  for (int e = 0; e < 16; ++e) {
    const int j = e >> 1, h = e & 1;
    const int k = (j < 4) ? (8 * g + 2 * j + h) : (16 + 8 * g + 2 * (j - 4) + h);
    a[e] = p[k];
  }
  return a;
}

// B fragment: 32x16 f16 tile stored [K][N] in LDS; lane holds column N = nBase + (lane&15),
// K striping same as frag_a.
__device__ __forceinline__ v16h frag_b(const _Float16* sh, int ldStride,
                                       int nBase, int kBase, int lane) {
  const int g = lane >> 4;
  const _Float16* p = sh + (size_t)kBase * ldStride + nBase + (lane & 15);
  v16h b;
#pragma unroll
  for (int e = 0; e < 16; ++e) {
    const int k = (e < 8) ? (8 * g + e) : (16 + 8 * g + (e - 8));
    b[e] = p[(size_t)k * ldStride];
  }
  return b;
}

#define WMMA_F16(a, b, c) \
  __builtin_amdgcn_wmma_f32_16x16x32_f16(false, (a), false, (b), (short)0, (c), false, false)

// Async global->LDS 128-bit copy (CDNA5 GLOBAL_LOAD_ASYNC_TO_LDS_B128, GVS mode).
// ldsaddr: LDS byte offset in a VGPR; voff: 32-bit byte offset; base: 64-bit SGPR base.
__device__ __forceinline__ void async_g2l_b128(unsigned ldsaddr, unsigned voff,
                                               const void* base) {
  asm volatile("global_load_async_to_lds_b128 %0, %1, %2"
               :: "v"(ldsaddr), "v"(voff), "s"((unsigned long long)base)
               : "memory");
}
__device__ __forceinline__ void wait_asynccnt0() {
  asm volatile("s_wait_asynccnt 0x0" ::: "memory");
}

// ---------------------------------------------------------------------------
// Kernel 1: QKV projection.  C[4096,1024] = X(f32) @ W(f32) + bias,
// output f16 in head-split layout [B, H, S, DEPTH].
// Block tile 128x128, K-tile 64 (2 WMMA k-steps per barrier window),
// 8 waves (4 in M x 2 in N), 16 WMMAs per wave per K-tile.
// ---------------------------------------------------------------------------
__global__ void __launch_bounds__(256)
qkv_proj_kernel(const float* __restrict__ X, const float* __restrict__ W,
                const float* __restrict__ bias, _Float16* __restrict__ Out) {
  __shared__ _Float16 Ash[128][72];   // 128 x 64 (+8 pad)
  __shared__ _Float16 Bsh[64][136];   // 64 x 128 (+8 pad)

  const int tid  = threadIdx.x;
  const int lane = tid & 31;
  const int wave = tid >> 5;
  const int wm   = (wave & 3) * 32;   // 4 waves along M, 32 rows each
  const int wn   = (wave >> 2) * 64;  // 2 waves along N, 64 cols each
  const int blockM = blockIdx.y * 128;
  const int blockN = blockIdx.x * 128;

  v8f acc[2][4];
#pragma unroll
  for (int i = 0; i < 2; ++i)
#pragma unroll
    for (int j = 0; j < 4; ++j) acc[i][j] = (v8f){};

  for (int k0 = 0; k0 < D_MODEL; k0 += 64) {
    {                                   // A tile: 128x64 f32 -> f16 (32/thread)
      const int r  = tid >> 1;
      const int c0 = (tid & 1) * 32;
      const float* src = X + (size_t)(blockM + r) * D_MODEL + k0 + c0;
#pragma unroll
      for (int c = 0; c < 32; ++c) Ash[r][c0 + c] = (_Float16)src[c];
    }
    {                                   // B tile: 64x128 f32 -> f16 (32/thread)
      const int r  = tid >> 2;
      const int c0 = (tid & 3) * 32;
      const float* src = W + (size_t)(k0 + r) * D_MODEL + blockN + c0;
#pragma unroll
      for (int c = 0; c < 32; ++c) Bsh[r][c0 + c] = (_Float16)src[c];
    }
    if (k0 + 64 < D_MODEL)              // prefetch next weight K-tile
      __builtin_prefetch(W + (size_t)(k0 + 64 + (tid >> 2)) * D_MODEL + blockN, 0, 1);
    __syncthreads();

#pragma unroll
    for (int ks = 0; ks < 64; ks += 32) {
      v16h af[2], bf[4];
#pragma unroll
      for (int mt = 0; mt < 2; ++mt) af[mt] = frag_a(&Ash[0][0], 72, wm + mt * 16, ks, lane);
#pragma unroll
      for (int nt = 0; nt < 4; ++nt) bf[nt] = frag_b(&Bsh[0][0], 136, wn + nt * 16, ks, lane);
#pragma unroll
      for (int mt = 0; mt < 2; ++mt)
#pragma unroll
        for (int nt = 0; nt < 4; ++nt) acc[mt][nt] = WMMA_F16(af[mt], bf[nt], acc[mt][nt]);
    }
    __syncthreads();
  }

  // Epilogue: C/D layout m = r + 8*(lane>>4), n = lane&15. Head-split store.
  const int g = lane >> 4, nloc = lane & 15;
#pragma unroll
  for (int mt = 0; mt < 2; ++mt)
#pragma unroll
    for (int nt = 0; nt < 4; ++nt)
#pragma unroll
      for (int r = 0; r < 8; ++r) {
        const int m = blockM + wm + mt * 16 + r + 8 * g;
        const int n = blockN + wn + nt * 16 + nloc;
        const float val = acc[mt][nt][r] + bias[n];
        const int b = m / SEQ, s = m % SEQ;
        const int h = n / DEPTH, d = n % DEPTH;
        Out[(((size_t)b * NUM_HEADS + h) * SEQ + s) * DEPTH + d] = (_Float16)val;
      }
}

// ---------------------------------------------------------------------------
// Kernel 2: logits = Qh @ Kh^T * (1/8) + mask*NEG_INF   -> attn buffer (raw)
// One WG per (bh, 64 q-rows); stream Sk in tiles of 128. K tile is stored
// TRANSPOSED into LDS ([depth][sk]) so frag_b reads it as a [K][N] tile.
// ---------------------------------------------------------------------------
__global__ void __launch_bounds__(256)
logits_kernel(const _Float16* __restrict__ Qh, const _Float16* __restrict__ Kh,
              const int* __restrict__ mask, float* __restrict__ attn) {
  const int bh = blockIdx.y;
  const int b  = bh / NUM_HEADS;
  const int q0 = blockIdx.x * 64;

  __shared__ _Float16 Qsh[64][72];     // 64 q x 64 depth (+8)
  __shared__ _Float16 KshT[64][136];   // 64 depth x 128 sk (+8), transposed

  const _Float16* Qbase = Qh + ((size_t)bh * SEQ + q0) * DEPTH;
  const _Float16* Kbase = Kh + (size_t)bh * SEQ * DEPTH;
  float* attnBase = attn + ((size_t)bh * SEQ + q0) * SEQ;

  const int tid = threadIdx.x, lane = tid & 31, wave = tid >> 5;
  {                                     // Q tile 64x64 (16/thread)
    const int r  = tid >> 2;
    const int c0 = (tid & 3) * 16;
    const _Float16* src = Qbase + (size_t)r * DEPTH + c0;
#pragma unroll
    for (int c = 0; c < 16; ++c) Qsh[r][c0 + c] = src[c];
  }
  __syncthreads();

  const int wrow = (wave & 3) * 16;     // 4 waves along q
  const int wcol = (wave >> 2) * 64;    // 2 waves along sk cols

  for (int s0 = 0; s0 < SEQ; s0 += 128) {
    {                                   // K tile 128x64, transpose into KshT
      const int r  = tid >> 1;          // sk row 0..127
      const int c0 = (tid & 1) * 32;    // depth cols
      const _Float16* src = Kbase + (size_t)(s0 + r) * DEPTH + c0;
#pragma unroll
      for (int c = 0; c < 32; ++c) KshT[c0 + c][r] = src[c];
    }
    __syncthreads();

    v8f acc[4];
#pragma unroll
    for (int nt = 0; nt < 4; ++nt) acc[nt] = (v8f){};

#pragma unroll
    for (int ks = 0; ks < DEPTH; ks += 32) {
      v16h a = frag_a(&Qsh[0][0], 72, wrow, ks, lane);
#pragma unroll
      for (int nt = 0; nt < 4; ++nt) {
        // B = Kh^T: B[k][n] = KshT[k][n]; lane = n, elem = k, strided.
        v16h bfr = frag_b(&KshT[0][0], 136, wcol + nt * 16, ks, lane);
        acc[nt] = WMMA_F16(a, bfr, acc[nt]);
      }
    }

    const int g = lane >> 4, nloc = lane & 15;
#pragma unroll
    for (int nt = 0; nt < 4; ++nt)
#pragma unroll
      for (int r = 0; r < 8; ++r) {
        const int qrow = wrow + r + 8 * g;
        const int col  = s0 + wcol + nt * 16 + nloc;
        const float lg = acc[nt][r] * 0.125f + (float)mask[b * SEQ + col] * NEG_INF;
        attnBase[(size_t)qrow * SEQ + col] = lg;
      }
    __syncthreads();
  }
}

// ---------------------------------------------------------------------------
// Kernel 3: row softmax, in place. One wave32 per row of 2048 (64 elems/lane).
// ---------------------------------------------------------------------------
__global__ void __launch_bounds__(256)
softmax_kernel(float* __restrict__ attn) {
  const size_t row = (size_t)blockIdx.x * 8 + (threadIdx.x >> 5);
  const int lane = threadIdx.x & 31;
  float* p = attn + row * SEQ;

  float vals[64];
  float m = -__builtin_inff();
#pragma unroll
  for (int i = 0; i < 64; ++i) {
    vals[i] = p[lane + i * 32];
    m = fmaxf(m, vals[i]);
  }
#pragma unroll
  for (int off = 16; off > 0; off >>= 1) m = fmaxf(m, __shfl_xor(m, off, 32));

  float sum = 0.f;
#pragma unroll
  for (int i = 0; i < 64; ++i) {
    vals[i] = __expf(vals[i] - m);
    sum += vals[i];
  }
#pragma unroll
  for (int off = 16; off > 0; off >>= 1) sum += __shfl_xor(sum, off, 32);

  const float inv = 1.0f / sum;
#pragma unroll
  for (int i = 0; i < 64; ++i) p[lane + i * 32] = vals[i] * inv;
}

// ---------------------------------------------------------------------------
// Kernel 4: out_h = attn @ Vh  ->  concat f16 [B, S, D_MODEL]
// WG tile: 128 q-rows x 64 depth, K-loop over Sk=2048 in steps of 32.
// V tile (pure f16 copy) is fetched with GLOBAL_LOAD_ASYNC_TO_LDS_B128:
// 256 lanes x 16B == one 32x64 f16 tile, drained with s_wait_asynccnt.
// ---------------------------------------------------------------------------
__global__ void __launch_bounds__(256)
av_kernel(const float* __restrict__ attn, const _Float16* __restrict__ Vh,
          _Float16* __restrict__ Concat) {
  const int bh = blockIdx.x;
  const int b = bh / NUM_HEADS, h = bh % NUM_HEADS;
  const int q0 = blockIdx.y * 128;

  __shared__ _Float16 Ash[128][40];   // 128 x 32 (+8)
  __shared__ _Float16 Bsh[32][72];    // 32 x 64 (+8)

  const float* Abase = attn + ((size_t)bh * SEQ + q0) * SEQ;
  const _Float16* Vbase = Vh + (size_t)bh * SEQ * DEPTH;

  const int tid = threadIdx.x, lane = tid & 31, wave = tid >> 5;

  // Per-thread async-copy geometry for the V tile (32 rows x 64 cols f16):
  // thread t -> row t>>3, col (t&7)*8 halves == 16 bytes.
  const int vrow = tid >> 3;
  const int vcol = (tid & 7) * 8;
  const unsigned vLds = (unsigned)(size_t)&Bsh[vrow][vcol];

  v8f acc[4];
#pragma unroll
  for (int nt = 0; nt < 4; ++nt) acc[nt] = (v8f){};

  for (int k0 = 0; k0 < SEQ; k0 += 32) {
    // V tile: async DMA into LDS (no VGPR round-trip, ASYNCcnt tracked)
    const unsigned vOff = (unsigned)(((size_t)(k0 + vrow) * DEPTH + vcol) * sizeof(_Float16));
    async_g2l_b128(vLds, vOff, (const void*)Vbase);

    {                                   // attn tile 128x32 f32 -> f16 (16/thread)
      const int r  = tid >> 1;
      const int c0 = (tid & 1) * 16;
      const float* src = Abase + (size_t)r * SEQ + k0 + c0;
#pragma unroll
      for (int c = 0; c < 16; ++c) Ash[r][c0 + c] = (_Float16)src[c];
    }
    wait_asynccnt0();                   // own async copy landed in LDS
    __syncthreads();

    v16h a = frag_a(&Ash[0][0], 40, wave * 16, 0, lane);
#pragma unroll
    for (int nt = 0; nt < 4; ++nt) {
      v16h bf = frag_b(&Bsh[0][0], 72, nt * 16, 0, lane);
      acc[nt] = WMMA_F16(a, bf, acc[nt]);
    }
    __syncthreads();
  }

  const int g = lane >> 4, nloc = lane & 15;
#pragma unroll
  for (int nt = 0; nt < 4; ++nt)
#pragma unroll
    for (int r = 0; r < 8; ++r) {
      const int qrow = q0 + wave * 16 + r + 8 * g;
      const int d = nt * 16 + nloc;
      Concat[((size_t)b * SEQ + qrow) * D_MODEL + h * DEPTH + d] = (_Float16)acc[nt][r];
    }
}

// ---------------------------------------------------------------------------
// Kernel 5: final projection.  out[4096,1024] = concat(f16) @ Wo + bo  (f32)
// Same tiling as qkv_proj, f16 A input, plain row-major f32 output.
// ---------------------------------------------------------------------------
__global__ void __launch_bounds__(256)
out_proj_kernel(const _Float16* __restrict__ X, const float* __restrict__ W,
                const float* __restrict__ bias, float* __restrict__ Out) {
  __shared__ _Float16 Ash[128][72];
  __shared__ _Float16 Bsh[64][136];

  const int tid  = threadIdx.x;
  const int lane = tid & 31;
  const int wave = tid >> 5;
  const int wm   = (wave & 3) * 32;
  const int wn   = (wave >> 2) * 64;
  const int blockM = blockIdx.y * 128;
  const int blockN = blockIdx.x * 128;

  v8f acc[2][4];
#pragma unroll
  for (int i = 0; i < 2; ++i)
#pragma unroll
    for (int j = 0; j < 4; ++j) acc[i][j] = (v8f){};

  for (int k0 = 0; k0 < D_MODEL; k0 += 64) {
    {                                   // A tile: 128x64 f16 (32/thread)
      const int r  = tid >> 1;
      const int c0 = (tid & 1) * 32;
      const _Float16* src = X + (size_t)(blockM + r) * D_MODEL + k0 + c0;
#pragma unroll
      for (int c = 0; c < 32; ++c) Ash[r][c0 + c] = src[c];
    }
    {                                   // B tile: 64x128 f32 -> f16 (32/thread)
      const int r  = tid >> 2;
      const int c0 = (tid & 3) * 32;
      const float* src = W + (size_t)(k0 + r) * D_MODEL + blockN + c0;
#pragma unroll
      for (int c = 0; c < 32; ++c) Bsh[r][c0 + c] = (_Float16)src[c];
    }
    if (k0 + 64 < D_MODEL)
      __builtin_prefetch(W + (size_t)(k0 + 64 + (tid >> 2)) * D_MODEL + blockN, 0, 1);
    __syncthreads();

#pragma unroll
    for (int ks = 0; ks < 64; ks += 32) {
      v16h af[2], bf[4];
#pragma unroll
      for (int mt = 0; mt < 2; ++mt) af[mt] = frag_a(&Ash[0][0], 72, wm + mt * 16, ks, lane);
#pragma unroll
      for (int nt = 0; nt < 4; ++nt) bf[nt] = frag_b(&Bsh[0][0], 136, wn + nt * 16, ks, lane);
#pragma unroll
      for (int mt = 0; mt < 2; ++mt)
#pragma unroll
        for (int nt = 0; nt < 4; ++nt) acc[mt][nt] = WMMA_F16(af[mt], bf[nt], acc[mt][nt]);
    }
    __syncthreads();
  }

  const int g = lane >> 4, nloc = lane & 15;
#pragma unroll
  for (int mt = 0; mt < 2; ++mt)
#pragma unroll
    for (int nt = 0; nt < 4; ++nt)
#pragma unroll
      for (int r = 0; r < 8; ++r) {
        const int m = blockM + wm + mt * 16 + r + 8 * g;
        const int n = blockN + wn + nt * 16 + nloc;
        Out[(size_t)m * D_MODEL + n] = acc[mt][nt][r] + bias[n];
      }
}

// ---------------------------------------------------------------------------
// Launch
// ---------------------------------------------------------------------------
extern "C" void kernel_launch(void* const* d_in, const int* in_sizes, int n_in,
                              void* d_out, int out_size, void* d_ws, size_t ws_size,
                              hipStream_t stream) {
  const float* v_in = (const float*)d_in[0];
  const float* k_in = (const float*)d_in[1];
  const float* q_in = (const float*)d_in[2];
  const int*   mask = (const int*)d_in[3];
  const float* Wq = (const float*)d_in[4];
  const float* bq = (const float*)d_in[5];
  const float* Wk = (const float*)d_in[6];
  const float* bk = (const float*)d_in[7];
  const float* Wv = (const float*)d_in[8];
  const float* bv = (const float*)d_in[9];
  const float* Wo = (const float*)d_in[10];
  const float* bo = (const float*)d_in[11];

  float* out  = (float*)d_out;                          // [B, S, D_MODEL]
  float* attn = out + (size_t)BATCH * SEQ * D_MODEL;    // [B, H, S, S]

  const size_t NE = (size_t)BATCH * NUM_HEADS * SEQ * DEPTH;  // 4,194,304
  _Float16* ws = (_Float16*)d_ws;
  _Float16* Qh = ws;
  _Float16* Kh = ws + NE;
  _Float16* Vh = ws + 2 * NE;
  _Float16* Cc = ws + 3 * NE;   // concat [B, S, D_MODEL]

  dim3 blk(256);
  dim3 gproj(D_MODEL / 128, (BATCH * SEQ) / 128);       // (8, 32)

  qkv_proj_kernel<<<gproj, blk, 0, stream>>>(q_in, Wq, bq, Qh);
  qkv_proj_kernel<<<gproj, blk, 0, stream>>>(k_in, Wk, bk, Kh);
  qkv_proj_kernel<<<gproj, blk, 0, stream>>>(v_in, Wv, bv, Vh);

  logits_kernel<<<dim3(SEQ / 64, BATCH * NUM_HEADS), blk, 0, stream>>>(Qh, Kh, mask, attn);

  softmax_kernel<<<dim3((BATCH * NUM_HEADS * SEQ) / 8), blk, 0, stream>>>(attn);

  av_kernel<<<dim3(BATCH * NUM_HEADS, SEQ / 128), blk, 0, stream>>>(attn, Vh, Cc);

  out_proj_kernel<<<gproj, blk, 0, stream>>>(Cc, Wo, bo, out);
}
